// GCN_15745350107951
// MI455X (gfx1250) — compile-verified, matrix operand order checked by reading, not run on hardware
//
#include <hip/hip_runtime.h>
#include <hip/hip_bf16.h>

typedef _Float16 half_t;
typedef __attribute__((ext_vector_type(16))) _Float16 v16h;
typedef __attribute__((ext_vector_type(8)))  _Float16 v8h;
typedef __attribute__((ext_vector_type(8)))  float    v8f;

// Problem constants (from reference setup_inputs)
#define BB   128   // batch
#define CC   16    // channels
#define NN   128   // nodes
#define EMB  16    // embedding
#define EE   4     // edge types
#define OUT  4     // EMB / EE

// ---------------------------------------------------------------------------
// Stage 0: An^T[e][m][n] = A[e][n][m] / max(||A[e][:,m]||_2, 1e-12), as f16.
// One block per (e,m) column; 128 threads reduce the squared column.
// ---------------------------------------------------------------------------
__global__ __launch_bounds__(128) void gcn_norm_transpose(
    const float* __restrict__ A, half_t* __restrict__ AnT) {
  const int em = blockIdx.x;          // e*128 + m
  const int e  = em >> 7;
  const int m  = em & 127;
  const int n  = threadIdx.x;         // row index within column

  __shared__ float red[128];
  const float a = A[(e * NN + n) * NN + m];
  red[n] = a * a;
  __syncthreads();
  #pragma unroll
  for (int s = 64; s > 0; s >>= 1) {
    if (n < s) red[n] += red[n + s];
    __syncthreads();
  }
  const float inv = 1.0f / fmaxf(sqrtf(red[0]), 1e-12f);
  AnT[(e * NN + m) * NN + n] = (half_t)(a * inv);
}

// ---------------------------------------------------------------------------
// Stage 1+2 fused. One workgroup handles 4 (b,c) pairs; 8 waves, each wave
// owns one 16-node tile.  Stage 1: Y = x @ Wcat via one padded-K WMMA per
// (bc, tile), Y^T kept in LDS as f16 [4 bc][16 j][128 node].  Stage 2: per
// edge e, 4 K-chunk WMMAs of An[e]^T (global f16) x B, where B's 16 columns
// are columns j=4e..4e+3 of the 4 LDS Y matrices (no wasted WMMA lanes).
// ---------------------------------------------------------------------------
__global__ __launch_bounds__(256) void gcn_fused(
    const float*  __restrict__ x,    // [B,C,N,EMB]
    const float*  __restrict__ W,    // [E,EMB,OUT]
    const half_t* __restrict__ AnT,  // [E,N(m),N(n)] f16
    float*        __restrict__ out)  // [B,E*C,N,OUT]
{
  __shared__ half_t Ylds[4 * 16 * NN];   // 16 KB

  const int tid  = threadIdx.x;
  const int wave = tid >> 5;             // 0..7  -> node tile
  const int lane = tid & 31;
  const int ln   = lane & 15;            // tile row (A) / tile col (B,C,D)
  const int lh   = lane >> 4;            // half-wave select
  const int bc0  = blockIdx.x * 4;       // first (b*C+c) pair of this block

  // ---- Stage 1: B fragment = Wcat[f][j], K=f (16, zero-padded to 32) ----
  v16h bw = {};
  {
    const int e = ln >> 2, o = ln & 3;                 // column j = e*4+o
    const float* wp = W + e * (EMB * OUT) + o;         // stride OUT over f
    #pragma unroll
    for (int i = 0; i < 8; ++i)                        // f = lh*8 + i
      bw[i] = (half_t)wp[(lh * 8 + i) * OUT];
  }

  #pragma unroll
  for (int g = 0; g < 4; ++g) {
    const int bc   = bc0 + g;
    const int node = wave * 16 + ln;                   // A-matrix row
    // A fragment: x[bc, node, f], f = lh*8 + 0..7 ; K 16..31 zero-padded
    const float4* xp = (const float4*)(x + ((size_t)bc * NN + node) * EMB + lh * 8);
    const float4 x0 = xp[0], x1 = xp[1];
    v16h ax = {};
    ax[0] = (half_t)x0.x; ax[1] = (half_t)x0.y;
    ax[2] = (half_t)x0.z; ax[3] = (half_t)x0.w;
    ax[4] = (half_t)x1.x; ax[5] = (half_t)x1.y;
    ax[6] = (half_t)x1.z; ax[7] = (half_t)x1.w;

    v8f acc = {};
    acc = __builtin_amdgcn_wmma_f32_16x16x32_f16(
        false, ax, false, bw, (short)0, acc, false, false);

    // D layout: VGPR r -> row M = lh*8 + r, lane col = ln (= j).
    // Store Y^T[g][j][node] : contiguous 8 halves per lane.
    v8h pk;
    #pragma unroll
    for (int r = 0; r < 8; ++r) pk[r] = (half_t)acc[r];
    *(v8h*)&Ylds[((g * 16 + ln) * NN) + wave * 16 + lh * 8] = pk;
  }
  __syncthreads();

  // ---- Stage 2: out tile = relu(An[e]^T @ Ypacked) ----
  const int m = wave * 16 + ln;          // A-matrix row (node)
  const int g  = ln >> 2;                // which bc pair this column serves
  const int oo = ln & 3;                 // output feature o
  const int bc = bc0 + g;
  const int b  = bc >> 4;                // bc / C
  const int c  = bc & 15;                // bc % C

  #pragma unroll
  for (int e = 0; e < EE; ++e) {
    v8f acc = {};
    const int jj = e * 4 + oo;           // Y column feeding this D column
    #pragma unroll
    for (int kc = 0; kc < 4; ++kc) {
      // A fragment: AnT[e][m][k], k = kc*32 + lh*8 + {0..7, 16..23}
      const half_t* ap = AnT + ((size_t)(e * NN + m) * NN + kc * 32 + lh * 8);
      const v8h a0 = *(const v8h*)ap;
      const v8h a1 = *(const v8h*)(ap + 16);
      const v16h af = __builtin_shufflevector(a0, a1,
          0,1,2,3,4,5,6,7,8,9,10,11,12,13,14,15);
      // B fragment from LDS: Y^T[g][jj][k], same k pattern
      const half_t* bp = &Ylds[(g * 16 + jj) * NN + kc * 32 + lh * 8];
      const v8h b0 = *(const v8h*)bp;
      const v8h b1 = *(const v8h*)(bp + 16);
      const v16h bf = __builtin_shufflevector(b0, b1,
          0,1,2,3,4,5,6,7,8,9,10,11,12,13,14,15);

      acc = __builtin_amdgcn_wmma_f32_16x16x32_f16(
          false, af, false, bf, (short)0, acc, false, false);
    }
    // relu + store: out[b][e*C + c][node][o]
    const int chan = e * CC + c;
    float* op = out + (((size_t)(b * (EE * CC) + chan) * NN) + wave * 16 + lh * 8) * OUT + oo;
    #pragma unroll
    for (int r = 0; r < 8; ++r)
      op[r * OUT] = fmaxf(acc[r], 0.0f);
  }
}

// ---------------------------------------------------------------------------
extern "C" void kernel_launch(void* const* d_in, const int* in_sizes, int n_in,
                              void* d_out, int out_size, void* d_ws, size_t ws_size,
                              hipStream_t stream) {
  const float* x = (const float*)d_in[0];   // [128,16,128,16]
  const float* A = (const float*)d_in[1];   // [4,128,128]
  const float* W = (const float*)d_in[2];   // [4,16,4]
  float* out = (float*)d_out;               // [128,64,128,4]
  half_t* AnT = (half_t*)d_ws;              // 4*128*128 f16 = 128 KB scratch

  gcn_norm_transpose<<<EE * NN, 128, 0, stream>>>(A, AnT);
  gcn_fused<<<(BB * CC) / 4, 256, 0, stream>>>(x, W, AnT, out);
}
